// GNNNet_16492674417057
// MI455X (gfx1250) — matrix operand order repaired and checked by reading, not compiled, not run on hardware
//
#include <hip/hip_runtime.h>
#include <math.h>

#define DF 128

typedef __attribute__((ext_vector_type(16))) __bf16 v16bf;
typedef __attribute__((ext_vector_type(8)))  float  v8f;

// ---------------- degree / norm precompute ----------------

__global__ __launch_bounds__(256) void k_init_deg(float* deg, int n) {
  int i = blockIdx.x * 256 + threadIdx.x;
  if (i < n) deg[i] = 1.0f;               // self-loop contributes 1
}

__global__ __launch_bounds__(256) void k_count_deg(const int* __restrict__ col,
                                                   float* __restrict__ deg, int ne) {
  int e = blockIdx.x * 256 + threadIdx.x;
  if (e < ne) atomicAdd(&deg[col[e]], 1.0f);
}

__global__ __launch_bounds__(256) void k_dinv(float* deg, int n) {
  int i = blockIdx.x * 256 + threadIdx.x;
  if (i < n) deg[i] = rsqrtf(deg[i]);     // deg >= 1 always (self-loops)
}

__global__ __launch_bounds__(256) void k_norm(const int* __restrict__ row,
                                              const int* __restrict__ col,
                                              const float* __restrict__ dinv,
                                              float* __restrict__ norm, int ne) {
  int e = blockIdx.x * 256 + threadIdx.x;
  if (e < ne) norm[e] = dinv[row[e]] * dinv[col[e]];
}

// ---------------- WMMA GEMM: HW = A(lda) @ W ----------------
// W split once per block into transposed bf16 hi/lo LDS planes [col][k].
// Each WAVE owns a distinct 16-row m-tile: A is loaded + split exactly once
// per tile (ah[4]/al[4] in VGPRs), then the wave sweeps the 8 column tiles,
// fetching B hi/lo fragments as contiguous 32B LDS loads (no conversion).
// 2-term bf16 split: A@B ~= Ah@Bh + Ah@Bl + Al@Bh, fp32 accumulate.

__global__ __launch_bounds__(256) void k_gemm(const float* __restrict__ A, long lda,
                                              const float* __restrict__ W,
                                              float* __restrict__ HW,
                                              int n, int mTiles) {
  __shared__ alignas(32) __bf16 sWhi[DF * DF];   // 32 KB: W^T high part
  __shared__ alignas(32) __bf16 sWlo[DF * DF];   // 32 KB: W^T low  part

  // cooperative transpose + split (once per block)
  for (int i = threadIdx.x; i < DF * DF; i += 256) {
    int k    = i >> 7;       // W row (K index)
    int ncol = i & (DF - 1); // W col (N index)
    float w = W[i];
    __bf16 wh = (__bf16)w;
    sWhi[ncol * DF + k] = wh;
    sWlo[ncol * DF + k] = (__bf16)(w - (float)wh);
  }
  __syncthreads();

  const int lane = threadIdx.x & 31;
  const int wv   = threadIdx.x >> 5;     // wave id 0..7 -> m-tile within block
  const int lr   = lane & 15;
  const int hi   = lane >> 4;            // lane half selects K sub-range

  const int stride = gridDim.x * 8;      // m-tiles advanced per grid sweep
  for (int mt = blockIdx.x * 8 + wv; mt < mTiles; mt += stride) {
    int mRow = mt * 16 + lr;
    if (mRow >= n) mRow = n - 1;         // tail clamp (stores are guarded)
    const float* arow = A + (long)mRow * lda;

    // prefetch this wave's next m-tile row (global_prefetch_b8)
    if (mt + stride < mTiles)
      __builtin_prefetch(arow + (long)stride * 16 * lda, 0, 0);

    // Load + split the whole 16x128 A tile ONCE per wave.
    // A k order chosen here (contiguous): lanes 0-15: elems 0..15 -> K=k0+0..15,
    // lanes 16-31 shifted by +16 (matches B).
    v16bf ah[4], al[4];
#pragma unroll
    for (int kk = 0; kk < 4; ++kk) {
      const int k0 = kk * 32 + hi * 16;
#pragma unroll
      for (int j = 0; j < 8; ++j) {
        float a0 = arow[k0 + j];
        __bf16 h0 = (__bf16)a0;
        ah[kk][j] = h0;
        al[kk][j] = (__bf16)(a0 - (float)h0);
        float a1 = arow[k0 + 8 + j];
        __bf16 h1 = (__bf16)a1;
        ah[kk][8 + j] = h1;
        al[kk][8 + j] = (__bf16)(a1 - (float)h1);
      }
    }

    const bool fullTile = (mt * 16 + 16) <= n;   // wave-uniform fast path
    const int  rBase    = mt * 16 + hi * 8;

#pragma unroll 1                          // keep rolled: don't pin 128 B VGPRs
    for (int nt = 0; nt < 8; ++nt) {
      const int nCol = nt * 16 + lr;
      v8f acc = {};
#pragma unroll
      for (int kk = 0; kk < 4; ++kk) {
        // B k order chosen here (same as A): lanes 0-15 hold K=kb..kb+15, lanes 16-31 hold +16;
        // W^T in LDS makes this 32 contiguous bytes per lane.
        const int kb = kk * 32 + hi * 16;
        v16bf bh = *reinterpret_cast<const v16bf*>(&sWhi[nCol * DF + kb]);
        v16bf bl = *reinterpret_cast<const v16bf*>(&sWlo[nCol * DF + kb]);
        acc = __builtin_amdgcn_wmma_f32_16x16x32_bf16(false, ah[kk], false, bh,
                                                      (short)0, acc, false, false);
        acc = __builtin_amdgcn_wmma_f32_16x16x32_bf16(false, ah[kk], false, bl,
                                                      (short)0, acc, false, false);
        acc = __builtin_amdgcn_wmma_f32_16x16x32_bf16(false, al[kk], false, bh,
                                                      (short)0, acc, false, false);
      }
      // C/D layout: VGPR v, lanes 0-15 -> (M=v, N=lane); lanes 16-31 -> (M=v+8).
      float* outp = HW + (long)rBase * DF + nCol;
      if (fullTile) {
        // uniform branch: straight global_store_b32 with immediate offsets
#pragma unroll
        for (int v = 0; v < 8; ++v) outp[v * DF] = acc[v];
      } else {
#pragma unroll
        for (int v = 0; v < 8; ++v)
          if (rBase + v < n) outp[v * DF] = acc[v];
      }
    }
  }
}

// ---------------- message passing ----------------

__global__ __launch_bounds__(256) void k_zero_slice(float* __restrict__ slice, int n) {
  long t = (long)blockIdx.x * 256 + threadIdx.x;
  int i = (int)(t >> 5);
  if (i >= n) return;
  int c = ((int)t & 31) * 4;
  float4 z = make_float4(0.f, 0.f, 0.f, 0.f);
  *reinterpret_cast<float4*>(slice + (long)i * (3 * DF) + c) = z;
}

// one wave per edge: each lane moves 4 floats (float4 gather + 4 fp32 atomics)
__global__ __launch_bounds__(256) void k_scatter(const float* __restrict__ hw,
                                                 const int* __restrict__ row,
                                                 const int* __restrict__ col,
                                                 const float* __restrict__ norm,
                                                 float* __restrict__ slice, int ne) {
  long t = (long)blockIdx.x * 256 + threadIdx.x;
  int e = (int)(t >> 5);
  if (e >= ne) return;
  int c = ((int)t & 31) * 4;
  float nm = norm[e];
  const float4 v = *reinterpret_cast<const float4*>(hw + (long)row[e] * DF + c);
  float* dst = slice + (long)col[e] * (3 * DF) + c;
  atomicAdd(dst + 0, nm * v.x);
  atomicAdd(dst + 1, nm * v.y);
  atomicAdd(dst + 2, nm * v.z);
  atomicAdd(dst + 3, nm * v.w);
}

// out = tanh(scattered + dinv^2 * hw (self-loop) + bias)
__global__ __launch_bounds__(256) void k_finalize(float* __restrict__ slice,
                                                  const float* __restrict__ hw,
                                                  const float* __restrict__ dinv,
                                                  const float* __restrict__ bias, int n) {
  long t = (long)blockIdx.x * 256 + threadIdx.x;
  int i = (int)(t >> 5);
  if (i >= n) return;
  int c = ((int)t & 31) * 4;
  float di = dinv[i];
  float s = di * di;
  const float4 hv = *reinterpret_cast<const float4*>(hw + (long)i * DF + c);
  const float4 bb = *reinterpret_cast<const float4*>(bias + c);
  float* o = slice + (long)i * (3 * DF) + c;
  o[0] = tanhf(o[0] + s * hv.x + bb.x);
  o[1] = tanhf(o[1] + s * hv.y + bb.y);
  o[2] = tanhf(o[2] + s * hv.z + bb.z);
  o[3] = tanhf(o[3] + s * hv.w + bb.w);
}

// ---------------- driver ----------------

extern "C" void kernel_launch(void* const* d_in, const int* in_sizes, int n_in,
                              void* d_out, int out_size, void* d_ws, size_t ws_size,
                              hipStream_t stream) {
  const float* x  = (const float*)d_in[0];
  const int*   ei = (const int*)d_in[1];
  const float* Ws[3] = {(const float*)d_in[2], (const float*)d_in[4], (const float*)d_in[6]};
  const float* bs[3] = {(const float*)d_in[3], (const float*)d_in[5], (const float*)d_in[7]};

  const int n  = in_sizes[0] / DF;   // 50000
  const int ne = in_sizes[1] / 2;    // 800000
  const int* row = ei;               // edge_index[0]
  const int* col = ei + ne;          // edge_index[1]
  float* out = (float*)d_out;        // [n, 3, DF]

  float* ws   = (float*)d_ws;
  float* deg  = ws;                                   // n floats (becomes dinv)
  float* norm = deg + (((long)n + 255) & ~255L);      // ne floats
  float* hw   = norm + (((long)ne + 255) & ~255L);    // n*DF floats

  dim3 blk(256);
  k_init_deg<<<(n + 255) / 256, blk, 0, stream>>>(deg, n);
  k_count_deg<<<(ne + 255) / 256, blk, 0, stream>>>(col, deg, ne);
  k_dinv<<<(n + 255) / 256, blk, 0, stream>>>(deg, n);
  k_norm<<<(ne + 255) / 256, blk, 0, stream>>>(row, col, deg, norm, ne);

  const int mTiles = (n + 15) / 16;
  const int featBlocksN = (n * 32 + 255) / 256;
  const int featBlocksE = (int)(((long)ne * 32 + 255) / 256);

  for (int l = 0; l < 3; ++l) {
    const float* A = (l == 0) ? x : (out + (long)(l - 1) * DF);
    long lda = (l == 0) ? DF : 3 * DF;
    k_gemm<<<dim3(512), blk, 0, stream>>>(A, lda, Ws[l], hw, n, mTiles);

    float* slice = out + (long)l * DF;
    k_zero_slice<<<featBlocksN, blk, 0, stream>>>(slice, n);
    k_scatter<<<featBlocksE, blk, 0, stream>>>(hw, row, col, norm, slice, ne);
    k_finalize<<<featBlocksN, blk, 0, stream>>>(slice, hw, deg, bs[l], n);
  }
}
